// FineMatcher_14663018349179
// MI455X (gfx1250) — matrix operand order repaired and checked by reading, compiled, not run on hardware
//
#include <hip/hip_runtime.h>

// FineMatcher for MI455X (gfx1250, wave32).
// One wave per match: stage 25x128 f32 -> f16 tile in LDS, 8x
// v_wmma_f32_16x16x32_f16 (B = mid broadcast into all 16 columns, so every
// D column holds the 25-score vector), then wave32 shfl softmax/expectation.

typedef __attribute__((ext_vector_type(16))) _Float16 v16h;
typedef __attribute__((ext_vector_type(8)))  _Float16 v8h;
typedef __attribute__((ext_vector_type(4)))  _Float16 v4h;
typedef __attribute__((ext_vector_type(8)))  float    v8f;
typedef __attribute__((ext_vector_type(4)))  float    v4f;   // clang ext-vector (NT-load friendly)

#define WAVES_PER_BLOCK 8
#define W2 25
#define C  128
#define ROWSTRIDE 136   // 128 halves + 8 pad halves -> 272B row stride (conflict-free b128)

__global__ __launch_bounds__(256) void FineMatcher_wmma_kernel(
    const float* __restrict__ f0,      // (M, 25, 128) f32
    const float* __restrict__ f1,      // (M, 25, 128) f32
    const float* __restrict__ mk,      // (M, 2) f32
    const int*   __restrict__ scale_p, // scalar
    float*       __restrict__ out,     // (M, 2) f32
    int M)
{
    __shared__ __align__(32) _Float16 sF1[WAVES_PER_BLOCK][W2 * ROWSTRIDE];
    __shared__ __align__(32) _Float16 sMid[WAVES_PER_BLOCK][C];
    __shared__ __align__(16) float    sSc[WAVES_PER_BLOCK][32];

    const int lane = threadIdx.x & 31;
    const int wv   = threadIdx.x >> 5;
    const int m    = blockIdx.x * WAVES_PER_BLOCK + wv;
    if (m >= M) return;  // uniform per wave -> EXEC stays all-ones for WMMA

    // ---- stage feat_f1[m] (25x128 f32 = 3200 floats) into LDS as f16 ----
    // iteration i covers exactly row i; lane loads float4 at column lane*4.
    const v4f* pf1 = (const v4f*)(f1 + (size_t)m * (W2 * C));
    #pragma unroll
    for (int i = 0; i < W2; ++i) {
        v4f v = __builtin_nontemporal_load(&pf1[i * 32 + lane]);
        v4h h = { (_Float16)v.x, (_Float16)v.y, (_Float16)v.z, (_Float16)v.w };
        *(v4h*)&sF1[wv][i * ROWSTRIDE + lane * 4] = h;
    }
    // ---- stage mid = feat_f0[m][12][:] (128 f32) ----
    {
        const v4f* pf0 = (const v4f*)(f0 + (size_t)m * (W2 * C) + 12 * C);
        v4f v = __builtin_nontemporal_load(&pf0[lane]);
        v4h h = { (_Float16)v.x, (_Float16)v.y, (_Float16)v.z, (_Float16)v.w };
        *(v4h*)&sMid[wv][lane * 4] = h;
    }
    asm volatile("s_wait_dscnt 0" ::: "memory");  // same-wave LDS RAW fence

    // ---- scores via WMMA: D = A(16x32 f16) x B(32x16 f16) + C(f32) ----
    // A 16-bit layout: lanes 0-15 (M=lane)  half h -> K = (h<8 ? h : h+8)
    //                  lanes 16-31 (M=lane-16)      -> K = (h<8 ? h+8 : h+16)
    // -> two aligned 16B LDS loads per fragment.
    // B: every column = mid chunk; same K order as A (lanes 0-15 K 0..7,16..23; lanes 16-31 K 8..15,24..31).
    v8f acc0 = {}; // rows 0..15
    v8f acc1 = {}; // rows 16..24 (25..31 zero-padded)
    const int klo  = (lane < 16) ? 0 : 8;   // A fragment K base (halves)
    const int bko  = (lane < 16) ? 0 : 16;  // unused (B loads use klo)
    const int row0 = lane & 15;
    const int row1 = 16 + row0;
    #pragma unroll
    for (int kc = 0; kc < 4; ++kc) {
        const int kb = kc * 32;
        v8h blo = *(const v8h*)&sMid[wv][kb + klo];
        v8h bhi = *(const v8h*)&sMid[wv][kb + klo + 16];
        v16h b = __builtin_shufflevector(blo, bhi,
                    0,1,2,3,4,5,6,7,8,9,10,11,12,13,14,15);

        v8h a0lo = *(const v8h*)&sF1[wv][row0 * ROWSTRIDE + kb + klo];
        v8h a0hi = *(const v8h*)&sF1[wv][row0 * ROWSTRIDE + kb + klo + 16];
        v16h a0 = __builtin_shufflevector(a0lo, a0hi,
                    0,1,2,3,4,5,6,7,8,9,10,11,12,13,14,15);

        v8h a1lo = {}; v8h a1hi = {};
        if (row1 < W2) {
            a1lo = *(const v8h*)&sF1[wv][row1 * ROWSTRIDE + kb + klo];
            a1hi = *(const v8h*)&sF1[wv][row1 * ROWSTRIDE + kb + klo + 16];
        }
        v16h a1 = __builtin_shufflevector(a1lo, a1hi,
                    0,1,2,3,4,5,6,7,8,9,10,11,12,13,14,15);

        acc0 = __builtin_amdgcn_wmma_f32_16x16x32_f16(
                   false, a0, false, b, (short)0, acc0, false, false);
        acc1 = __builtin_amdgcn_wmma_f32_16x16x32_f16(
                   false, a1, false, b, (short)0, acc1, false, false);
    }

    // ---- gather the 25 scores (D layout: lane j<16 = col j, VGPR r = row r;
    //      lanes 16-31 = rows r+8; all columns identical) ----
    if (lane == 0) {
        #pragma unroll
        for (int r = 0; r < 8; ++r) {
            sSc[wv][r]      = acc0[r];   // rows 0..7
            sSc[wv][16 + r] = acc1[r];   // rows 16..23
        }
    } else if (lane == 16) {
        #pragma unroll
        for (int r = 0; r < 8; ++r)
            sSc[wv][8 + r]  = acc0[r];   // rows 8..15
        sSc[wv][24] = acc1[0];           // row 24
    }
    asm volatile("s_wait_dscnt 0" ::: "memory");

    // ---- softmax over 25 + expected grid position (wave32 shfl reductions) ----
    const float inv_sqrtC = 0.08838834764831845f; // 1/sqrt(128)
    float v = (lane < W2) ? sSc[wv][lane] * inv_sqrtC : -3.0e38f;
    float mx = v;
    #pragma unroll
    for (int off = 16; off; off >>= 1)
        mx = fmaxf(mx, __shfl_xor(mx, off, 32));
    float e  = (lane < W2) ? __expf(v - mx) : 0.0f;
    float gx = (float)(lane / 5) * 0.5f - 1.0f;  // pos[i] = i*0.5 - 1
    float gy = (float)(lane % 5) * 0.5f - 1.0f;
    float s = e, sx = gx * e, sy = gy * e;
    #pragma unroll
    for (int off = 16; off; off >>= 1) {
        s  += __shfl_xor(s,  off, 32);
        sx += __shfl_xor(sx, off, 32);
        sy += __shfl_xor(sy, off, 32);
    }
    if (lane == 0) {
        float fct = 2.0f * (float)scale_p[0];     // (window//2) * scale
        float inv = 1.0f / s;
        float2 pt = ((const float2*)mk)[m];
        float2 o;
        o.x = pt.x + sx * inv * fct;
        o.y = pt.y + sy * inv * fct;
        ((float2*)out)[m] = o;
    }
}

extern "C" void kernel_launch(void* const* d_in, const int* in_sizes, int n_in,
                              void* d_out, int out_size, void* d_ws, size_t ws_size,
                              hipStream_t stream) {
    const float* f0 = (const float*)d_in[0];
    const float* f1 = (const float*)d_in[1];
    const float* mk = (const float*)d_in[2];
    const int*   sc = (const int*)d_in[3];
    float* out = (float*)d_out;

    const int M = in_sizes[0] / (W2 * C);
    const int blocks = (M + WAVES_PER_BLOCK - 1) / WAVES_PER_BLOCK;
    FineMatcher_wmma_kernel<<<blocks, 256, 0, stream>>>(f0, f1, mk, sc, out, M);
}
